// LocalEquiConv_71356586656065
// MI455X (gfx1250) — compile-verified, matrix operand order checked by reading, not run on hardware
//
#include <hip/hip_runtime.h>

typedef __attribute__((ext_vector_type(16))) _Float16 v16h;
typedef __attribute__((ext_vector_type(8)))  _Float16 v8h;
typedef __attribute__((ext_vector_type(8)))  float    v8f;
typedef __attribute__((ext_vector_type(4)))  float    v4f;
typedef __attribute__((ext_vector_type(4)))  unsigned int u32x4;
typedef __attribute__((ext_vector_type(8)))  int      i32x8;
typedef __attribute__((ext_vector_type(4)))  int      i32x4;

#define N_NODES   10000
#define NE        160000
#define FEA_ROW   80
#define N_TILES   10000

// ---- workspace byte offsets ----
#define B_W1    0u        // 2048 f16  (W1 pack: K=32, NT=4)
#define B_W2    4096u     // 4096 f16  (K=64, NT=4)
#define B_W3    12288u    // 4096 f16
#define B_BIAS  20480u    // 192 f32 (b1|b2|b3)
#define B_WSS   24576u    // 49152 f16 (K=32, N=1536)
#define B_WVV   122880u   // 24576 f16 (K=32, N=768)
#define B_WSV   172032u   // 8192  f16 (K=32, N=256)
#define B_WVS   188416u   // 16384 f16 (K=32, N=512)
#define B_ACC   221184u   // node accumulators (f32)
#define ACC_SI  0
#define ACC_SJ  160000
#define ACC_VI  320000
#define ACC_VJ  800000
#define ACC_TOTAL 1280000

#if defined(__has_builtin)
#if __has_builtin(__builtin_amdgcn_tensor_load_to_lds)
#define HAVE_TDM 1
#else
#define HAVE_TDM 0
#endif
#else
#define HAVE_TDM 0
#endif

static __device__ __forceinline__ int a_kof(int hi, int t) {
  // CDNA5 16-bit A/B fragment K mapping: lanes 0-15 hold K {0..7,16..23},
  // lanes 16-31 hold K {8..15,24..31}
  return (t < 8) ? (8 * hi + t) : (16 + 8 * hi + (t - 8));
}

static __device__ __forceinline__ v8f wmma32(v16h a, v16h b, v8f c) {
  return __builtin_amdgcn_wmma_f32_16x16x32_f16(false, a, false, b, (short)0, c,
                                                false, false);
}

static __device__ __forceinline__ v16h b_frag(const _Float16* pack, int blk, int lane) {
  // pre-swizzled blocks: 32 lanes x 16 halves each
  return *(const v16h*)(pack + (size_t)blk * 512 + lane * 16);
}

static __device__ __forceinline__ v16h a_frag_f32(const float* row, int hi) {
  v4f p0 = *(const v4f*)(row + 8 * hi);
  v4f p1 = *(const v4f*)(row + 8 * hi + 4);
  v4f p2 = *(const v4f*)(row + 16 + 8 * hi);
  v4f p3 = *(const v4f*)(row + 16 + 8 * hi + 4);
  v16h a;
#pragma unroll
  for (int t = 0; t < 4; ++t) {
    a[t]      = (_Float16)p0[t];
    a[4 + t]  = (_Float16)p1[t];
    a[8 + t]  = (_Float16)p2[t];
    a[12 + t] = (_Float16)p3[t];
  }
  return a;
}

static __device__ __forceinline__ v16h a_frag_f16(const _Float16* row, int hi) {
  v8h lo = *(const v8h*)(row + 8 * hi);
  v8h hh = *(const v8h*)(row + 16 + 8 * hi);
  v16h a;
#pragma unroll
  for (int t = 0; t < 8; ++t) { a[t] = lo[t]; a[8 + t] = hh[t]; }
  return a;
}

// fast activations: v_exp_f32 + v_rcp_f32 (avoid IEEE div expansion)
static __device__ __forceinline__ float sigm_f(float x) {
  return __builtin_amdgcn_rcpf(1.f + __expf(-x));
}
static __device__ __forceinline__ float silu_f(float x) { return x * sigm_f(x); }
static __device__ __forceinline__ void wavefence() { asm volatile("" ::: "memory"); }
static __device__ __forceinline__ void atomAddF(float* p, float v) {
  __hip_atomic_fetch_add(p, v, __ATOMIC_RELAXED, __HIP_MEMORY_SCOPE_AGENT);
}

#if HAVE_TDM
// 1-D TDM copy: n8 elements of 8 bytes, global -> LDS offset lds_off.
// D# per CDNA5 ISA §8.3/8.4: group0 {count=1, lds_addr, global_addr, type=2},
// group1 {data_size=3 (8B), tensor_dim0=n8, tensor_dim1=1, tile_dim0=n8,
//         tensor_dim0_stride=n8}, groups 2/3 zero (<=2D tensor).
static __device__ __forceinline__ void tdm_load_to_lds(unsigned lds_off,
                                                       const void* gsrc,
                                                       unsigned n8) {
  unsigned long long ga = (unsigned long long)gsrc;
  u32x4 g0;
  g0[0] = 1u;                                                  // count=1
  g0[1] = lds_off;                                             // lds_addr
  g0[2] = (unsigned)ga;                                        // global lo
  g0[3] = (unsigned)((ga >> 32) & 0x1FFFFFFu) | (2u << 30);    // global hi|type=2
  i32x8 g1;
  g1[0] = (int)(3u << 16);                       // wg_mask=0, data_size=8B
  g1[1] = (int)((n8 & 0xFFFFu) << 16);           // tensor_dim0[15:0]
  g1[2] = (int)((n8 >> 16) | (1u << 16));        // tensor_dim0[31:16]|tensor_dim1=1
  g1[3] = (int)((n8 & 0xFFFFu) << 16);           // tile_dim0 = n8
  g1[4] = 0;                                     // tile_dim1=0, tile_dim2=0
  g1[5] = (int)n8;                               // tensor_dim0_stride[31:0]
  g1[6] = 0;
  g1[7] = 0;
  i32x4 z4 = {0, 0, 0, 0};
#if __clang_major__ >= 23
  i32x8 z8 = {0, 0, 0, 0, 0, 0, 0, 0};
  __builtin_amdgcn_tensor_load_to_lds(g0, g1, z4, z4, z8, 0);
#else
  __builtin_amdgcn_tensor_load_to_lds(g0, g1, z4, z4, 0);
#endif
}
#endif

// ======================= kernel 0: repack weights, zero accumulators =========
__global__ void prep_kernel(const float* __restrict__ w1, const float* __restrict__ b1,
                            const float* __restrict__ w2, const float* __restrict__ b2,
                            const float* __restrict__ w3, const float* __restrict__ b3,
                            const float* __restrict__ w_ss, const float* __restrict__ w_vv,
                            const float* __restrict__ w_sv, const float* __restrict__ w_vs,
                            char* __restrict__ ws) {
  const int tid = blockIdx.x * blockDim.x + threadIdx.x;
  const int stride = gridDim.x * blockDim.x;

  float* bias = (float*)(ws + B_BIAS);
  for (int p = tid; p < 192; p += stride)
    bias[p] = (p < 64) ? b1[p] : ((p < 128) ? b2[p - 64] : b3[p - 128]);

  float* acc = (float*)(ws + B_ACC);
  for (int p = tid; p < ACC_TOTAL; p += stride) acc[p] = 0.f;

  const float s_ss = 0.022097086912079608f;  // (1/sqrt2)/32
  const float s_vv = 0.018042195912175804f;  // (1/sqrt2)*(1/sqrt3)/sqrt(512)
  const float s_sv = 0.03125f;               // (1/sqrt2)/sqrt(512) = 1/32
  const float s_vs = 0.022097086912079608f;  // (1/sqrt2)/32

  {  // W1: K=32 (KS=1), NT=4 ; B[k][n] = w1[k*64+n]
    _Float16* dst = (_Float16*)(ws + B_W1);
    for (int p = tid; p < 4 * 512; p += stride) {
      int blk = p >> 9, w = p & 511, lane = w >> 4, t = w & 15;
      int k = a_kof(lane >> 4, t);
      int n = blk * 16 + (lane & 15);
      dst[p] = (_Float16)w1[k * 64 + n];
    }
  }
  {  // W2/W3: K=64 (KS=2), NT=4
    _Float16* d2 = (_Float16*)(ws + B_W2);
    _Float16* d3 = (_Float16*)(ws + B_W3);
    for (int p = tid; p < 8 * 512; p += stride) {
      int blk = p >> 9, w = p & 511, lane = w >> 4, t = w & 15;
      int ks = blk >> 2, nt = blk & 3;
      int k = ks * 32 + a_kof(lane >> 4, t);
      int n = nt * 16 + (lane & 15);
      d2[p] = (_Float16)w2[k * 64 + n];
      d3[p] = (_Float16)w3[k * 64 + n];
    }
  }
  {  // Wss: K = j (32), n = i*48+o ; B[j][n] = w_ss[o,i,j]*s_ss
    _Float16* dst = (_Float16*)(ws + B_WSS);
    for (int p = tid; p < 96 * 512; p += stride) {
      int blk = p >> 9, w = p & 511, lane = w >> 4, t = w & 15;
      int k = a_kof(lane >> 4, t);
      int n = blk * 16 + (lane & 15);
      int i = n / 48, o = n % 48;
      dst[p] = (_Float16)(w_ss[((size_t)o * 32 + i) * 32 + k] * s_ss);
    }
  }
  {  // Wvv: K = i (32), n = j*48+o ; B[i][n] = w_vv[o,i,j]*s_vv
    _Float16* dst = (_Float16*)(ws + B_WVV);
    for (int p = tid; p < 48 * 512; p += stride) {
      int blk = p >> 9, w = p & 511, lane = w >> 4, t = w & 15;
      int k = a_kof(lane >> 4, t);
      int n = blk * 16 + (lane & 15);
      int j = n / 48, o = n % 48;
      dst[p] = (_Float16)(w_vv[((size_t)o * 32 + k) * 16 + j] * s_vv);
    }
  }
  {  // Wsv: K = i (32), n = j*16+o ; B[i][n] = w_sv[o,i,j]*s_sv
    _Float16* dst = (_Float16*)(ws + B_WSV);
    for (int p = tid; p < 16 * 512; p += stride) {
      int blk = p >> 9, w = p & 511, lane = w >> 4, t = w & 15;
      int k = a_kof(lane >> 4, t);
      int n = blk * 16 + (lane & 15);
      int j = n / 16, o = n % 16;
      dst[p] = (_Float16)(w_sv[((size_t)o * 32 + k) * 16 + j] * s_sv);
    }
  }
  {  // Wvs: K = j (32), n = i*16+o ; B[j][n] = w_vs[o,i,j]*s_vs
    _Float16* dst = (_Float16*)(ws + B_WVS);
    for (int p = tid; p < 32 * 512; p += stride) {
      int blk = p >> 9, w = p & 511, lane = w >> 4, t = w & 15;
      int k = a_kof(lane >> 4, t);
      int n = blk * 16 + (lane & 15);
      int i = n / 16, o = n % 16;
      dst[p] = (_Float16)(w_vs[((size_t)o * 32 + i) * 32 + k] * s_vs);
    }
  }
}

// ======================= kernel 1: WMMA MLP + mix + scatter ==================
#define K1_WAVES 8
#define K1_TPW 4
#define K1_STG_BASE 21504
#define K1_STG_SZ 2560

__global__ void mlp_scatter_kernel(const float* __restrict__ edge_fea,
                                   const float* __restrict__ edge_sh,
                                   const int* __restrict__ edge_index,
                                   char* __restrict__ ws) {
  extern __shared__ char smem[];
  const int tid = threadIdx.x;
#if HAVE_TDM
  if ((tid >> 5) == 0) {  // one wave issues the TDM weight DMA (21248 B = 2656 x 8B)
    tdm_load_to_lds(0u, ws, 21248u / 8u);
    __builtin_amdgcn_s_wait_tensorcnt(0);
  }
  __syncthreads();
#else
  {
    const v4f* src = (const v4f*)ws;
    v4f* dst = (v4f*)smem;
    for (int p = tid; p < 21248 / 16; p += blockDim.x) dst[p] = src[p];
  }
  __syncthreads();
#endif

  const int lane = tid & 31, wv = tid >> 5;
  const int hi = lane >> 4, ml = lane & 15;
  const _Float16* W1p = (const _Float16*)(smem + B_W1);
  const _Float16* W2p = (const _Float16*)(smem + B_W2);
  const _Float16* W3p = (const _Float16*)(smem + B_W3);
  const float* bias = (const float*)(smem + B_BIAS);
  char* stg = smem + K1_STG_BASE + wv * K1_STG_SZ;
  _Float16* hstage = (_Float16*)stg;      // 16x64 f16
  float* shT = (float*)(stg + 2048);      // 4x16 f32
  int* idxs = (int*)(stg + 2304);         // 16 i32

  float* accSi = (float*)(ws + B_ACC) + ACC_SI;
  float* accSj = (float*)(ws + B_ACC) + ACC_SJ;
  float* accVi = (float*)(ws + B_ACC) + ACC_VI;
  float* accVj = (float*)(ws + B_ACC) + ACC_VJ;

  const int wgid = blockIdx.x * K1_WAVES + wv;
  for (int it = 0; it < K1_TPW; ++it) {
    const int tile = wgid * K1_TPW + it;
    if (tile >= N_TILES) break;
    const int e = tile * 16 + ml;

    if (hi == 0) {
      v4f sh = *(const v4f*)(edge_sh + (size_t)e * 4);
      shT[ml] = sh.x; shT[16 + ml] = sh.y; shT[32 + ml] = sh.z; shT[48 + ml] = sh.w;
      idxs[ml] = edge_index[e];  // segment index is idx_i for all four scatters
    }

    v16h aS = a_frag_f32(edge_fea + (size_t)e * FEA_ROW, hi);
    // layer 1: 16x32 @ 32x64
#pragma unroll
    for (int nt = 0; nt < 4; ++nt) {
      v8f z = {};
      z = wmma32(aS, b_frag(W1p, nt, lane), z);
      float bn = bias[nt * 16 + ml];
#pragma unroll
      for (int r = 0; r < 8; ++r)
        hstage[(r + 8 * hi) * 64 + nt * 16 + ml] = (_Float16)silu_f(z[r] + bn);
    }
    wavefence();
    v16h h0 = a_frag_f16(hstage + ml * 64, hi);
    v16h h1 = a_frag_f16(hstage + ml * 64 + 32, hi);
    wavefence();
    // layer 2: 16x64 @ 64x64
#pragma unroll
    for (int nt = 0; nt < 4; ++nt) {
      v8f z = {};
      z = wmma32(h0, b_frag(W2p, nt, lane), z);
      z = wmma32(h1, b_frag(W2p, 4 + nt, lane), z);
      float bn = bias[64 + nt * 16 + ml];
#pragma unroll
      for (int r = 0; r < 8; ++r)
        hstage[(r + 8 * hi) * 64 + nt * 16 + ml] = (_Float16)silu_f(z[r] + bn);
    }
    wavefence();
    h0 = a_frag_f16(hstage + ml * 64, hi);
    h1 = a_frag_f16(hstage + ml * 64 + 32, hi);
    // layer 3 (no activation): "weight" = 16x64 tile in wt[]
    v8f wt[4];
#pragma unroll
    for (int nt = 0; nt < 4; ++nt) {
      v8f z = {};
      z = wmma32(h0, b_frag(W3p, nt, lane), z);
      z = wmma32(h1, b_frag(W3p, 4 + nt, lane), z);
      float bn = bias[128 + nt * 16 + ml];
#pragma unroll
      for (int r = 0; r < 8; ++r) z[r] += bn;
      wt[nt] = z;
    }
    wavefence();
    // mix with edge_sh, scatter-add into node accumulators at idx_i
#pragma unroll
    for (int r = 0; r < 8; ++r) {
      const int row = r + 8 * hi;
      const int nI = idxs[row];
      const float s0 = shT[row], s1 = shT[16 + row], s2 = shT[32 + row], s3 = shT[48 + row];
      atomAddF(accSi + (size_t)nI * 16 + ml, wt[0][r] * s0);
      atomAddF(accSj + (size_t)nI * 16 + ml, wt[2][r] * s0);
      float* vI = accVi + ((size_t)nI * 16 + ml) * 3;
      atomAddF(vI + 0, wt[1][r] * s1);
      atomAddF(vI + 1, wt[1][r] * s2);
      atomAddF(vI + 2, wt[1][r] * s3);
      float* vJ = accVj + ((size_t)nI * 16 + ml) * 3;
      atomAddF(vJ + 0, wt[3][r] * s1);
      atomAddF(vJ + 1, wt[3][r] * s2);
      atomAddF(vJ + 2, wt[3][r] * s3);
    }
    wavefence();
  }
}

// ======================= kernel 2: gather + fused bilinear WMMA + output =====
#define K2_WAVES 4
#define K2_TPW 4
#define K2_WB 196608
#define K2_STG 24576

__global__ void bilinear_kernel(const float* __restrict__ edge_fea,
                                const int* __restrict__ edge_index,
                                char* __restrict__ ws,
                                float* __restrict__ out) {
  extern __shared__ char smem[];
  const int tid = threadIdx.x;
#if HAVE_TDM
  if ((tid >> 5) == 0) {  // one wave DMAs all 192 KB of packed B matrices
    tdm_load_to_lds(0u, ws + B_WSS, K2_WB / 8u);
    __builtin_amdgcn_s_wait_tensorcnt(0);
  }
  __syncthreads();
#else
  {
    const v4f* src = (const v4f*)(ws + B_WSS);
    v4f* dst = (v4f*)smem;
    for (int p = tid; p < K2_WB / 16; p += blockDim.x) dst[p] = src[p];
  }
  __syncthreads();
#endif

  const int lane = tid & 31, wv = tid >> 5;
  const int hi = lane >> 4, ml = lane & 15;
  const _Float16* Wss = (const _Float16*)(smem);
  const _Float16* Wvv = (const _Float16*)(smem + (B_WVV - B_WSS));
  const _Float16* Wsv = (const _Float16*)(smem + (B_WSV - B_WSS));
  const _Float16* Wvs = (const _Float16*)(smem + (B_WVS - B_WSS));
  char* stg = smem + K2_WB + wv * K2_STG;
  float* S1row = (float*)stg;            // [m][i]    16x32
  float* S1T   = (float*)(stg + 2048);   // [i][m]    32x16
  float* V1row = (float*)(stg + 4096);   // [c][m][i] 3x16x32
  float* V1T   = (float*)(stg + 10240);  // [c][i][m] 3x32x16
  float* V2T   = (float*)(stg + 16384);  // [c][j][m] 3x16x16
  float* outs  = (float*)(stg + 19456);  // [m][80]

  const float* accSi = (const float*)(ws + B_ACC) + ACC_SI;
  const float* accSj = (const float*)(ws + B_ACC) + ACC_SJ;
  const float* accVi = (const float*)(ws + B_ACC) + ACC_VI;
  const float* accVj = (const float*)(ws + B_ACC) + ACC_VJ;

  const int wgid = blockIdx.x * K2_WAVES + wv;
  for (int it = 0; it < K2_TPW; ++it) {
    const int tile = wgid * K2_TPW + it;
    if (tile >= N_TILES) break;
    const int ebase = tile * 16;
    const int e = ebase + ml;
    const int nI = edge_index[e];
    const int nJ = edge_index[NE + e];

    // ---- gather/stage S1, V1, V2 (hi=0 lanes: i-half; hi=1 lanes: j-half)
    {
      const float* srcS = hi ? (accSj + (size_t)nJ * 16) : (accSi + (size_t)nI * 16);
#pragma unroll
      for (int q = 0; q < 4; ++q) {
        v4f v = *(const v4f*)(srcS + 4 * q);
#pragma unroll
        for (int u = 0; u < 4; ++u) {
          int i = 16 * hi + 4 * q + u;
          S1row[ml * 32 + i] = v[u];
          S1T[i * 16 + ml]   = v[u];
        }
      }
      const float* srcV = hi ? (accVj + (size_t)nJ * 48) : (accVi + (size_t)nI * 48);
#pragma unroll
      for (int q = 0; q < 12; ++q) {
        v4f v = *(const v4f*)(srcV + 4 * q);
#pragma unroll
        for (int u = 0; u < 4; ++u) {
          int t = 4 * q + u;
          int i = 16 * hi + t / 3, cc = t % 3;
          V1row[cc * 512 + ml * 32 + i] = v[u];
          V1T[cc * 512 + i * 16 + ml]   = v[u];
        }
      }
      const float* srcF = edge_fea + (size_t)e * FEA_ROW + 32 + 24 * hi;
#pragma unroll
      for (int q = 0; q < 6; ++q) {
        v4f v = *(const v4f*)(srcF + 4 * q);
#pragma unroll
        for (int u = 0; u < 4; ++u) {
          int t = 24 * hi + 4 * q + u;
          V2T[(t % 3) * 256 + (t / 3) * 16 + ml] = v[u];
        }
      }
    }
    wavefence();

    v16h aS2 = a_frag_f32(edge_fea + (size_t)e * FEA_ROW, hi);
    v16h aS1 = a_frag_f32(S1row + ml * 32, hi);

    v8f os0 = {}, os1 = {}, os2 = {};
    v8f ov0 = {}, ov1 = {}, ov2 = {};

    // M1: o_s += S1[,i] .* (S2 @ Wss)[, i*48+o]
#pragma unroll 4
    for (int i = 0; i < 32; ++i) {
      v8f s1v = *(const v8f*)(S1T + i * 16 + 8 * hi);
      v8f z = {};
      v8f c0 = wmma32(aS2, b_frag(Wss, i * 3 + 0, lane), z);
      v8f c1 = wmma32(aS2, b_frag(Wss, i * 3 + 1, lane), z);
      v8f c2 = wmma32(aS2, b_frag(Wss, i * 3 + 2, lane), z);
#pragma unroll
      for (int r = 0; r < 8; ++r) {
        os0[r] += s1v[r] * c0[r];
        os1[r] += s1v[r] * c1[r];
        os2[r] += s1v[r] * c2[r];
      }
    }
    // M2: o_s += sum_c V2c[,j] .* (V1c @ Wvv)[, j*48+o]
#pragma unroll
    for (int cc = 0; cc < 3; ++cc) {
      v16h aV = a_frag_f32(V1row + cc * 512 + ml * 32, hi);
#pragma unroll 4
      for (int j = 0; j < 16; ++j) {
        v8f v2v = *(const v8f*)(V2T + cc * 256 + j * 16 + 8 * hi);
        v8f z = {};
        v8f c0 = wmma32(aV, b_frag(Wvv, j * 3 + 0, lane), z);
        v8f c1 = wmma32(aV, b_frag(Wvv, j * 3 + 1, lane), z);
        v8f c2 = wmma32(aV, b_frag(Wvv, j * 3 + 2, lane), z);
#pragma unroll
        for (int r = 0; r < 8; ++r) {
          os0[r] += v2v[r] * c0[r];
          os1[r] += v2v[r] * c1[r];
          os2[r] += v2v[r] * c2[r];
        }
      }
    }
    // M3: o_v[c] += V2c[,j] .* (S1 @ Wsv)[, j*16+o]
#pragma unroll 4
    for (int j = 0; j < 16; ++j) {
      v8f z = {};
      v8f c0 = wmma32(aS1, b_frag(Wsv, j, lane), z);
      v8f a0 = *(const v8f*)(V2T + 0 * 256 + j * 16 + 8 * hi);
      v8f a1 = *(const v8f*)(V2T + 1 * 256 + j * 16 + 8 * hi);
      v8f a2 = *(const v8f*)(V2T + 2 * 256 + j * 16 + 8 * hi);
#pragma unroll
      for (int r = 0; r < 8; ++r) {
        ov0[r] += a0[r] * c0[r];
        ov1[r] += a1[r] * c0[r];
        ov2[r] += a2[r] * c0[r];
      }
    }
    // M4: o_v[c] += V1c[,i] .* (S2 @ Wvs)[, i*16+o]
#pragma unroll 4
    for (int i = 0; i < 32; ++i) {
      v8f z = {};
      v8f c0 = wmma32(aS2, b_frag(Wvs, i, lane), z);
      v8f a0 = *(const v8f*)(V1T + 0 * 512 + i * 16 + 8 * hi);
      v8f a1 = *(const v8f*)(V1T + 1 * 512 + i * 16 + 8 * hi);
      v8f a2 = *(const v8f*)(V1T + 2 * 512 + i * 16 + 8 * hi);
#pragma unroll
      for (int r = 0; r < 8; ++r) {
        ov0[r] += a0[r] * c0[r];
        ov1[r] += a1[r] * c0[r];
        ov2[r] += a2[r] * c0[r];
      }
    }
    // epilogue: silu / sigmoid gate (gate lane == o_v lane), stage + store
#pragma unroll
    for (int r = 0; r < 8; ++r) {
      const int row = r + 8 * hi;
      float g = sigm_f(os2[r]);
      outs[row * 80 + ml]      = silu_f(os0[r]);
      outs[row * 80 + 16 + ml] = silu_f(os1[r]);
      outs[row * 80 + 32 + ml * 3 + 0] = ov0[r] * g;
      outs[row * 80 + 32 + ml * 3 + 1] = ov1[r] * g;
      outs[row * 80 + 32 + ml * 3 + 2] = ov2[r] * g;
    }
    wavefence();
    {
      v4f* dstg = (v4f*)(out + (size_t)ebase * 80);
      const v4f* s4 = (const v4f*)outs;
#pragma unroll
      for (int q = 0; q < 10; ++q) dstg[lane + 32 * q] = s4[lane + 32 * q];
    }
    wavefence();
  }
}

// ======================= host launcher =======================================
extern "C" void kernel_launch(void* const* d_in, const int* in_sizes, int n_in,
                              void* d_out, int out_size, void* d_ws, size_t ws_size,
                              hipStream_t stream) {
  (void)in_sizes; (void)n_in; (void)out_size; (void)ws_size;
  const float* edge_sh    = (const float*)d_in[0];
  const float* edge_fea   = (const float*)d_in[1];
  const int*   edge_index = (const int*)d_in[2];
  const float* w1 = (const float*)d_in[4];
  const float* b1 = (const float*)d_in[5];
  const float* w2 = (const float*)d_in[6];
  const float* b2 = (const float*)d_in[7];
  const float* w3 = (const float*)d_in[8];
  const float* b3 = (const float*)d_in[9];
  const float* w_ss = (const float*)d_in[10];
  const float* w_vv = (const float*)d_in[11];
  const float* w_sv = (const float*)d_in[12];
  const float* w_vs = (const float*)d_in[13];
  char* ws = (char*)d_ws;
  float* out = (float*)d_out;

  prep_kernel<<<256, 256, 0, stream>>>(w1, b1, w2, b2, w3, b3,
                                       w_ss, w_vv, w_sv, w_vs, ws);
  // 313 blocks * 8 waves * 4 tiles >= 10000 tiles
  mlp_scatter_kernel<<<313, 256, K1_STG_BASE + K1_WAVES * K1_STG_SZ, stream>>>(
      edge_fea, edge_sh, edge_index, ws);
  // 625 blocks * 4 waves * 4 tiles == 10000 tiles
  bilinear_kernel<<<625, 128, K2_WB + K2_WAVES * K2_STG, stream>>>(
      edge_fea, edge_index, ws, out);
}
